// GConvGRUClassifier_73332271612041
// MI455X (gfx1250) — compile-verified, matrix operand order checked by reading, not run on hardware
//
#include <hip/hip_runtime.h>
#include <cstdint>
#include <cstddef>

typedef __attribute__((ext_vector_type(16))) _Float16 v16h;
typedef __attribute__((ext_vector_type(8)))  float    v8f;

#define T_STEPS 8
#define NNODES  100000
#define FDIM    64
#define HDIM    64
#define CDIM    16
#define NEDGES  1250000

__device__ inline void atomF(float* p, float v) {
  __hip_atomic_fetch_add(p, v, __ATOMIC_RELAXED, __HIP_MEMORY_SCOPE_AGENT);
}
__device__ inline float sigmoidf_(float x) { return 1.0f / (1.0f + __expf(-x)); }

// ---------------------------------------------------------------- utilities
__global__ void zerof_kernel(float* __restrict__ p, int n) {
  int i = blockIdx.x * blockDim.x + threadIdx.x;
  if (i < n) p[i] = 0.0f;
}

// deg[src[e]] += 1
__global__ void deg_kernel(const long long* __restrict__ src, float* __restrict__ deg, int e) {
  int i = blockIdx.x * blockDim.x + threadIdx.x;
  if (i < e) atomF(&deg[(int)src[i]], 1.0f);
}

__global__ void dinv_kernel(const float* __restrict__ deg, float* __restrict__ dinv, int n) {
  int i = blockIdx.x * blockDim.x + threadIdx.x;
  if (i < n) {
    float d = deg[i];
    dinv[i] = (d > 0.0f) ? rsqrtf(fmaxf(d, 1.0f)) : 0.0f;
  }
}

// w[e] = -dinv[src]*dinv[dst]; also int64 -> int32 index conversion
__global__ void edgew_kernel(const long long* __restrict__ ei, float* __restrict__ w,
                             int* __restrict__ s32, int* __restrict__ d32,
                             const float* __restrict__ dinv, int e) {
  int i = blockIdx.x * blockDim.x + threadIdx.x;
  if (i < e) {
    int s = (int)ei[i];
    int d = (int)ei[(size_t)e + i];
    s32[i] = s; d32[i] = d;
    w[i] = -dinv[s] * dinv[d];
  }
}

// out[dst] += w * x[src] ; one thread per (edge, 16-feature chunk)
__global__ void scatter_kernel(const float* __restrict__ x, const float* __restrict__ w,
                               const int* __restrict__ s32, const int* __restrict__ d32,
                               float* __restrict__ out, int e) {
  int idx = blockIdx.x * blockDim.x + threadIdx.x;
  if (idx >= e * 4) return;
  int ed = idx >> 2;
  int c  = (idx & 3) << 4;          // 0,16,32,48
  int s = s32[ed], d = d32[ed];
  float wv = w[ed];
  const float4* xp = (const float4*)(x + (size_t)s * 64 + c);
  float4 a = xp[0], b = xp[1], cc = xp[2], dd = xp[3];
  float* op = out + (size_t)d * 64 + c;
  atomF(op + 0,  wv * a.x);  atomF(op + 1,  wv * a.y);
  atomF(op + 2,  wv * a.z);  atomF(op + 3,  wv * a.w);
  atomF(op + 4,  wv * b.x);  atomF(op + 5,  wv * b.y);
  atomF(op + 6,  wv * b.z);  atomF(op + 7,  wv * b.w);
  atomF(op + 8,  wv * cc.x); atomF(op + 9,  wv * cc.y);
  atomF(op + 10, wv * cc.z); atomF(op + 11, wv * cc.w);
  atomF(op + 12, wv * dd.x); atomF(op + 13, wv * dd.y);
  atomF(op + 14, wv * dd.z); atomF(op + 15, wv * dd.w);
}

// ---------------------------------------------------------------- weight packing (row-major f16)
__global__ void pack_zr_kernel(const float* __restrict__ Wxz, const float* __restrict__ Whz,
                               const float* __restrict__ Wxr, const float* __restrict__ Whr,
                               const float* __restrict__ bxz, const float* __restrict__ bhz,
                               const float* __restrict__ bxr, const float* __restrict__ bhr,
                               _Float16* __restrict__ B, float* __restrict__ bias) {
  int idx = blockIdx.x * blockDim.x + threadIdx.x;
  if (idx < 256 * 128) {
    int k = idx >> 7, oc = idx & 127;
    int kb = k >> 6, i = k & 63, o = oc & 63;
    const float* Wx = (oc >= 64) ? Wxr : Wxz;
    const float* Wh = (oc >= 64) ? Whr : Whz;
    float v;
    if      (kb == 0) v = Wx[i * 64 + o];
    else if (kb == 1) v = Wx[4096 + i * 64 + o];
    else if (kb == 2) v = Wh[i * 64 + o];
    else              v = Wh[4096 + i * 64 + o];
    B[(size_t)k * 128 + oc] = (_Float16)v;
  }
  if (idx < 128) {
    int o = idx & 63;
    bias[idx] = (idx >= 64) ? (bxr[o] + bhr[o]) : (bxz[o] + bhz[o]);
  }
}

__global__ void pack_h_kernel(const float* __restrict__ Wxh, const float* __restrict__ Whh,
                              const float* __restrict__ bxh, const float* __restrict__ bhh,
                              _Float16* __restrict__ B, float* __restrict__ bias) {
  int idx = blockIdx.x * blockDim.x + threadIdx.x;
  if (idx < 256 * 64) {
    int k = idx >> 6, oc = idx & 63;
    int kb = k >> 6, i = k & 63;
    float v;
    if      (kb == 0) v = Wxh[i * 64 + oc];
    else if (kb == 1) v = Wxh[4096 + i * 64 + oc];
    else if (kb == 2) v = Whh[i * 64 + oc];
    else              v = Whh[4096 + i * 64 + oc];
    B[(size_t)k * 64 + oc] = (_Float16)v;
  }
  if (idx < 64) bias[idx] = bxh[idx] + bhh[idx];
}

__global__ void pack_lin_kernel(const float* __restrict__ Wlin, _Float16* __restrict__ B) {
  int idx = blockIdx.x * blockDim.x + threadIdx.x;
  if (idx < 64 * 16) B[idx] = (_Float16)Wlin[idx];
}

// ---------------------------------------------------------------- B fragment swizzle
// Row-major (K x OC) f16 -> fragment layout: frag f = kt*(OC/16)+n holds
// 32 lanes x 16 halves, lane-contiguous, matching v_wmma_f32_16x16x32_f16's
// per-lane B operand order (lane<16: K in {0-7,16-23}; lane>=16: {8-15,24-31}).
template <int OC>
__global__ void swizzle_kernel(const _Float16* __restrict__ Brm, _Float16* __restrict__ Bf, int K) {
  int idx = blockIdx.x * blockDim.x + threadIdx.x;
  if (idx >= K * OC) return;
  int e    = idx & 15;
  int lane = (idx >> 4) & 31;
  int f    = idx >> 9;
  constexpr int NT = OC / 16;
  int n  = f % NT;
  int kt = f / NT;
  bool hi = lane >= 16;
  int kk = (e < 8) ? (hi ? e + 8 : e) : (hi ? e + 16 : e + 8);
  int k   = kt * 32 + kk;
  int col = n * 16 + (lane & 15);
  Bf[idx] = Brm[(size_t)k * OC + col];
}

// ---------------------------------------------------------------- WMMA GEMM + fused GRU epilogues
// ACC(nrows x OC) = concat_cols[a0|a1|a2|a3](nrows x NSRC*64, f32 -> f16) @ Bf + bias
// EPI=0: out = ACC + bias                       (final linear)
// EPI=1: z = sigmoid(ACC_z), r = sigmoid(ACC_r); out=z, aux1 = aux0 * r   (OC=128)
// EPI=2: out = aux0*out + (1-aux0)*tanh(ACC)    (GRU h update in place, OC=64)
template <int NSRC, int OC, int EPI>
__global__ __launch_bounds__(256)
void wmma_gemm_kernel(const float* __restrict__ a0, const float* __restrict__ a1,
                      const float* __restrict__ a2, const float* __restrict__ a3,
                      const _Float16* __restrict__ Bf, const float* __restrict__ bias,
                      float* __restrict__ out, const float* __restrict__ aux0,
                      float* __restrict__ aux1, int nrows) {
  constexpr int K     = NSRC * 64;
  constexpr int NT    = OC / 16;
  constexpr int NELEM = K * OC;              // halves
  extern __shared__ _Float16 ldsB[];         // NELEM halves, fragment layout

  __builtin_prefetch(Bf, 0, 0);              // global_prefetch_b8: warm L2

  {
    const uint4* srcp = (const uint4*)Bf;
    uint4* dstp = (uint4*)ldsB;
    for (int i = threadIdx.x; i < NELEM / 8; i += blockDim.x) dstp[i] = srcp[i];
  }
  __syncthreads();

  const int lane  = threadIdx.x & 31;
  const int wave  = threadIdx.x >> 5;
  const int mtile = blockIdx.x * (blockDim.x >> 5) + wave;
  const int row   = mtile * 16 + (lane & 15);
  const bool hi   = (lane >= 16);
  const int col0  = lane & 15;

  const float* srcs[4] = {a0, a1, a2, a3};

  v8f acc[NT];
#pragma unroll
  for (int n = 0; n < NT; n++)
#pragma unroll
    for (int e = 0; e < 8; e++) acc[n][e] = 0.0f;

#pragma unroll
  for (int kt = 0; kt < K / 32; kt++) {
    const int k0 = kt * 32;
    const float* A = srcs[k0 >> 6];
    const int cb = k0 & 63;
    const int r0 = cb + (hi ? 8 : 0);
    const int r1 = cb + (hi ? 24 : 16);

    v16h af;
#pragma unroll
    for (int e = 0; e < 16; e++) af[e] = (_Float16)0.0f;
    if (row < nrows) {
      const float* base = A + (size_t)row * 64;
      float4 x0 = *(const float4*)(base + r0);
      float4 x1 = *(const float4*)(base + r0 + 4);
      float4 x2 = *(const float4*)(base + r1);
      float4 x3 = *(const float4*)(base + r1 + 4);
      af[0]  = (_Float16)x0.x; af[1]  = (_Float16)x0.y; af[2]  = (_Float16)x0.z; af[3]  = (_Float16)x0.w;
      af[4]  = (_Float16)x1.x; af[5]  = (_Float16)x1.y; af[6]  = (_Float16)x1.z; af[7]  = (_Float16)x1.w;
      af[8]  = (_Float16)x2.x; af[9]  = (_Float16)x2.y; af[10] = (_Float16)x2.z; af[11] = (_Float16)x2.w;
      af[12] = (_Float16)x3.x; af[13] = (_Float16)x3.y; af[14] = (_Float16)x3.z; af[15] = (_Float16)x3.w;
    }

#pragma unroll
    for (int n = 0; n < NT; n++) {
      // fragment-packed: one contiguous 32B chunk per lane -> 2x ds_load_b128
      const v16h bf = *(const v16h*)(ldsB + ((size_t)(kt * NT + n) * 512 + lane * 16));
      acc[n] = __builtin_amdgcn_wmma_f32_16x16x32_f16(
          false, af, false, bf, (short)0, acc[n], false, false);
    }
  }

  if (EPI == 0) {
#pragma unroll
    for (int n = 0; n < NT; n++) {
      const int col = n * 16 + col0;
      const float bv = bias[col];
#pragma unroll
      for (int e = 0; e < 8; e++) {
        const int m = mtile * 16 + e + (hi ? 8 : 0);
        if (m < nrows) out[(size_t)m * OC + col] = acc[n][e] + bv;
      }
    }
  } else if (EPI == 1) {
    // acc[0..3] -> z pre-act (cols 0-63), acc[4..7] -> r pre-act (cols 64-127)
#pragma unroll
    for (int n = 0; n < NT / 2; n++) {
      const int col = n * 16 + col0;
      const float bz = bias[col];
      const float br = bias[col + 64];
#pragma unroll
      for (int e = 0; e < 8; e++) {
        const int m = mtile * 16 + e + (hi ? 8 : 0);
        if (m < nrows) {
          const size_t off = (size_t)m * 64 + col;
          float zv = sigmoidf_(acc[n][e] + bz);
          float rv = sigmoidf_(acc[n + NT / 2][e] + br);
          out[off]  = zv;                 // z
          aux1[off] = aux0[off] * rv;     // hr = h * r
        }
      }
    }
  } else {
    // EPI == 2: h = z*h + (1-z)*tanh(acc + bias), in place on `out`
#pragma unroll
    for (int n = 0; n < NT; n++) {
      const int col = n * 16 + col0;
      const float bv = bias[col];
#pragma unroll
      for (int e = 0; e < 8; e++) {
        const int m = mtile * 16 + e + (hi ? 8 : 0);
        if (m < nrows) {
          const size_t off = (size_t)m * 64 + col;
          float ht = tanhf(acc[n][e] + bv);
          float zv = aux0[off];
          out[off] = zv * out[off] + (1.0f - zv) * ht;
        }
      }
    }
  }
}

// ---------------------------------------------------------------- launcher
static inline size_t align_up(size_t x) { return (x + 255) & ~(size_t)255; }

extern "C" void kernel_launch(void* const* d_in, const int* in_sizes, int n_in,
                              void* d_out, int out_size, void* d_ws, size_t ws_size,
                              hipStream_t stream) {
  const float*     X    = (const float*)d_in[0];
  const long long* EI   = (const long long*)d_in[1];
  const float* Wxz = (const float*)d_in[2];  const float* bxz = (const float*)d_in[3];
  const float* Whz = (const float*)d_in[4];  const float* bhz = (const float*)d_in[5];
  const float* Wxr = (const float*)d_in[6];  const float* bxr = (const float*)d_in[7];
  const float* Whr = (const float*)d_in[8];  const float* bhr = (const float*)d_in[9];
  const float* Wxh = (const float*)d_in[10]; const float* bxh = (const float*)d_in[11];
  const float* Whh = (const float*)d_in[12]; const float* bhh = (const float*)d_in[13];
  const float* Wlin = (const float*)d_in[14]; const float* blin = (const float*)d_in[15];
  float* OUT = (float*)d_out;

  const int N = NNODES, E = NEDGES;
  const int N64 = N * 64;

  // workspace carve-up
  char* p = (char*)d_ws;
  auto take = [&](size_t bytes) { char* r = p; p += align_up(bytes); return r; };
  float*    deg    = (float*)take((size_t)N * 4);
  float*    dinv   = (float*)take((size_t)N * 4);
  float*    w      = (float*)take((size_t)E * 4);
  int*      s32    = (int*)take((size_t)E * 4);
  int*      d32    = (int*)take((size_t)E * 4);
  _Float16* BzrRM  = (_Float16*)take((size_t)256 * 128 * 2);
  _Float16* BhRM   = (_Float16*)take((size_t)256 * 64 * 2);
  _Float16* BlinRM = (_Float16*)take((size_t)64 * 16 * 2);
  _Float16* Bzr    = (_Float16*)take((size_t)256 * 128 * 2);   // fragment-packed
  _Float16* Bh     = (_Float16*)take((size_t)256 * 64 * 2);
  _Float16* Blin   = (_Float16*)take((size_t)64 * 16 * 2);
  float*    bzr    = (float*)take(128 * 4);
  float*    bh     = (float*)take(64 * 4);
  float*    h      = (float*)take((size_t)N64 * 4);
  float*    mvx    = (float*)take((size_t)N64 * 4);
  float*    mvh    = (float*)take((size_t)N64 * 4);   // reused for mv(h*r)
  float*    hr     = (float*)take((size_t)N64 * 4);
  float*    zb     = (float*)take((size_t)N64 * 4);
  (void)ws_size; (void)n_in; (void)in_sizes; (void)out_size;

  const int TB = 256;
  auto blocks = [&](long long n) { return (int)((n + TB - 1) / TB); };

  // --- graph preprocessing ---
  zerof_kernel<<<blocks(N), TB, 0, stream>>>(deg, N);
  deg_kernel<<<blocks(E), TB, 0, stream>>>(EI, deg, E);
  dinv_kernel<<<blocks(N), TB, 0, stream>>>(deg, dinv, N);
  edgew_kernel<<<blocks(E), TB, 0, stream>>>(EI, w, s32, d32, dinv, E);

  // --- weight packing + fragment swizzle ---
  pack_zr_kernel<<<blocks(256 * 128), TB, 0, stream>>>(Wxz, Whz, Wxr, Whr, bxz, bhz, bxr, bhr, BzrRM, bzr);
  pack_h_kernel<<<blocks(256 * 64), TB, 0, stream>>>(Wxh, Whh, bxh, bhh, BhRM, bh);
  pack_lin_kernel<<<blocks(64 * 16), TB, 0, stream>>>(Wlin, BlinRM);
  swizzle_kernel<128><<<blocks(256 * 128), TB, 0, stream>>>(BzrRM, Bzr, 256);
  swizzle_kernel<64><<<blocks(256 * 64), TB, 0, stream>>>(BhRM, Bh, 256);
  swizzle_kernel<16><<<blocks(64 * 16), TB, 0, stream>>>(BlinRM, Blin, 64);

  // --- h0 = 0 ---
  zerof_kernel<<<blocks(N64), TB, 0, stream>>>(h, N64);

  const int mtiles = (N + 15) / 16;          // 6250
  const int gemm_blocks = (mtiles + 7) / 8;  // 8 waves / block

  for (int t = 0; t < T_STEPS; t++) {
    const float* xt = X + (size_t)t * N64;

    // mvx = L~ x_t
    zerof_kernel<<<blocks(N64), TB, 0, stream>>>(mvx, N64);
    scatter_kernel<<<blocks((long long)E * 4), TB, 0, stream>>>(xt, w, s32, d32, mvx, E);
    // mvh = L~ h
    zerof_kernel<<<blocks(N64), TB, 0, stream>>>(mvh, N64);
    scatter_kernel<<<blocks((long long)E * 4), TB, 0, stream>>>(h, w, s32, d32, mvh, E);

    // fused: z = sigmoid([xt|mvx|h|mvh]@Bzr + bzr)_z ; hr = h*sigmoid(..)_r
    wmma_gemm_kernel<4, 128, 1><<<gemm_blocks, 256, 256 * 128 * 2, stream>>>(
        xt, mvx, h, mvh, Bzr, bzr, zb, h, hr, N);

    // mvh (reused) = L~ (h*r)
    zerof_kernel<<<blocks(N64), TB, 0, stream>>>(mvh, N64);
    scatter_kernel<<<blocks((long long)E * 4), TB, 0, stream>>>(hr, w, s32, d32, mvh, E);

    // fused: h = z*h + (1-z)*tanh([xt|mvx|hr|mvhr]@Bh + bh)
    wmma_gemm_kernel<4, 64, 2><<<gemm_blocks, 256, 256 * 64 * 2, stream>>>(
        xt, mvx, hr, mvh, Bh, bh, h, zb, nullptr, N);
  }

  // OUT = h @ Blin + blin
  wmma_gemm_kernel<1, 16, 0><<<gemm_blocks, 256, 64 * 16 * 2, stream>>>(
      h, h, h, h, Blin, blin, OUT, nullptr, nullptr, N);
}
